// AllegroHaikuLayer_8933531975690
// MI455X (gfx1250) — matrix operand order repaired and checked by reading, not compiled, not run on hardware
//
#include <hip/hip_runtime.h>

typedef __attribute__((ext_vector_type(16))) __bf16 v16bf;
typedef __attribute__((ext_vector_type(8)))  __bf16 v8bf;
typedef __attribute__((ext_vector_type(8)))  float  v8f;
typedef __attribute__((address_space(3))) unsigned short lds_ushort;

#define N_EDGES 32768
#define N_NODES 2048
#define CG_TOTAL 573

// ---------------- bf16 helpers (RNE, bit-level) -----------------------------
__device__ inline unsigned short f2bfu(float f) {
  unsigned u = __builtin_bit_cast(unsigned, f);
  return (unsigned short)((u + 0x7FFFu + ((u >> 16) & 1u)) >> 16);
}

// A fragment: K(ev) = kb + 16*half + ev -> 16 contiguous
__device__ inline v16bf loadA_row(const unsigned short* row, int kb, int half) {
  v8bf lo = *(const v8bf*)(row + kb + 16 * half);
  v8bf hi = *(const v8bf*)(row + kb + 16 * half + 8);
  return __builtin_shufflevector(lo, hi, 0,1,2,3,4,5,6,7,8,9,10,11,12,13,14,15);
}
// B fragment from N-major bf16 weights: K(ev) = kb + 16*half + ev -> 16 contiguous
__device__ inline v16bf loadB_col(const unsigned short* col, int kb, int half) {
  const v8bf* p = (const v8bf*)(col + kb + 16 * half);
  v8bf lo = p[0], hi = p[1];
  return __builtin_shufflevector(lo, hi, 0,1,2,3,4,5,6,7,8,9,10,11,12,13,14,15);
}

// async global->LDS 16B copy (gfx1250 GLOBAL_LOAD_ASYNC_TO_LDS_B128, ASYNCcnt)
__device__ inline void async_cp16(unsigned lds_addr, unsigned long long gaddr) {
  asm volatile("global_load_async_to_lds_b128 %0, %1, off" :: "v"(lds_addr), "v"(gaddr) : "memory");
}
__device__ inline void async_wait0() {
  asm volatile("s_wait_asynccnt 0" ::: "memory");
}
__device__ inline unsigned lds_off_of(const void* p) {
  return (unsigned)(unsigned long long)(lds_ushort*)p;
}

// ---------------- CG path tables -------------------------------------------
__device__ __constant__ int c_pl1[13] = {0,0,0,1,1,1,1,2,2,2,2,3,3};
__device__ __constant__ int c_pl2[13] = {0,1,2,0,1,1,2,0,1,2,2,1,2};
__device__ __constant__ int c_pl3[13] = {0,1,2,1,0,2,1,2,1,0,2,2,1};
__device__ __constant__ int c_off[13] = {0,1,10,35,44,53,98,143,168,213,238,363,468};

// ---------------- device CG math (f64, mirrors reference) ------------------
__device__ double dfac(int n) { double r = 1.0; for (int i = 2; i <= n; ++i) r *= (double)i; return r; }

__device__ double su2cg(int j1,int m1,int j2,int m2,int j3,int m3) {
  if (m1 + m2 != m3) return 0.0;
  int vmin = 0;
  if (j2 - j3 - m1 > vmin) vmin = j2 - j3 - m1;
  if (j1 - j3 + m2 > vmin) vmin = j1 - j3 + m2;
  int vmax = j1 + j2 - j3;
  if (j1 - m1 < vmax) vmax = j1 - m1;
  if (j2 + m2 < vmax) vmax = j2 + m2;
  if (vmax < vmin) return 0.0;
  double pref = sqrt((2.0*j3+1.0)*dfac(j3+j1-j2)*dfac(j3-j1+j2)*dfac(j1+j2-j3)/dfac(j1+j2+j3+1));
  pref *= sqrt(dfac(j3+m3)*dfac(j3-m3)*dfac(j1-m1)*dfac(j1+m1)*dfac(j2-m2)*dfac(j2+m2));
  double s = 0.0;
  for (int v = vmin; v <= vmax; ++v) {
    double d = dfac(v)*dfac(j1+j2-j3-v)*dfac(j1-m1-v)*dfac(j2+m2-v)*dfac(j3-j2+m1+v)*dfac(j3-j1-m2+v);
    s += ((v & 1) ? -1.0 : 1.0) / d;
  }
  return pref * s;
}

struct D2 { double x, y; };
__device__ D2 Qent(int l, int a, int i) {
  int m = a - l;
  const double s = 0.70710678118654752440;
  double pm = (m & 1) ? -1.0 : 1.0;
  if (m < 0) {
    if (i == a)         return D2{0.0,  s};
    if (i == 2*l - a)   return D2{0.0, -s * pm};
    return D2{0.0, 0.0};
  } else if (m == 0) {
    return (i == l) ? D2{1.0, 0.0} : D2{0.0, 0.0};
  } else {
    if (i == 2*l - a)   return D2{s, 0.0};
    if (i == a)         return D2{s * pm, 0.0};
    return D2{0.0, 0.0};
  }
}

__global__ void cg_init_kernel(float* cg) {
  __shared__ double sre[CG_TOTAL];
  __shared__ double sim[CG_TOTAL];
  int g = threadIdx.x;
  int p = -1, loc = 0;
  if (g < CG_TOTAL) {
    for (int q = 12; q >= 0; --q) if (g >= c_off[q]) { p = q; loc = g - c_off[q]; break; }
    int l1 = c_pl1[p], l2 = c_pl2[p], l3 = c_pl3[p];
    int n2 = 2*l2 + 1, n3 = 2*l3 + 1;
    int a = loc / (n2 * n3); int rem = loc % (n2 * n3);
    int b = rem / n3;        int c = rem % n3;
    double re = 0.0, im = 0.0;
    for (int i = 0; i < 2*l1 + 1; ++i)
      for (int j = 0; j < n2; ++j) {
        int k = (i - l1) + (j - l2) + l3;
        if (k < 0 || k >= n3) continue;
        double cgv = su2cg(l1, i - l1, l2, j - l2, l3, k - l3);
        if (cgv == 0.0) continue;
        D2 q1 = Qent(l1, a, i), q2 = Qent(l2, b, j), q3 = Qent(l3, c, k);
        q3.y = -q3.y;
        D2 t; t.x = q1.x*q2.x - q1.y*q2.y; t.y = q1.x*q2.y + q1.y*q2.x;
        re += (t.x*q3.x - t.y*q3.y) * cgv;
        im += (t.x*q3.y + t.y*q3.x) * cgv;
      }
    sre[g] = re; sim[g] = im;
  }
  __syncthreads();
  if (g < CG_TOTAL) {
    int l3 = c_pl3[p];
    int sz = (2*c_pl1[p]+1) * (2*c_pl2[p]+1) * (2*l3+1);
    int o = c_off[p];
    double nr = 0.0, ni = 0.0;
    for (int t = 0; t < sz; ++t) { nr += sre[o+t]*sre[o+t]; ni += sim[o+t]*sim[o+t]; }
    double val = (nr >= ni) ? sre[g] / sqrt(nr) : sim[g] / sqrt(ni);
    val *= sqrt(2.0*l3 + 1.0);
    cg[g] = (float)val;
  }
}

// ---------------- zero scratch ---------------------------------------------
__global__ void zero_kernel(float* p, int n) {
  int i = blockIdx.x * blockDim.x + threadIdx.x;
  if (i < n) p[i] = 0.0f;
}

// ---------------- weight prep: transpose to N-major, scale, bf16 ----------
__global__ void prep_w_kernel(const float* __restrict__ W, unsigned short* __restrict__ Wt,
                              int K, float scale) {
  int idx = blockIdx.x * blockDim.x + threadIdx.x;
  if (idx >= K * 64) return;
  int n = idx / K, k = idx % K;
  Wt[idx] = f2bfu(W[k * 64 + n] * scale);
}

// ---------------- edge preprocess: Y, envelope, w, xbf, atomic scatter -----
__global__ void edge_pre_kernel(const float* __restrict__ vec, const float* __restrict__ x,
                                const int* __restrict__ senders, const float* __restrict__ W_w,
                                float* __restrict__ nodeWY, float* __restrict__ env,
                                unsigned short* __restrict__ xbf) {
  int e    = blockIdx.x * 8 + (threadIdx.x >> 5);
  int lane = threadIdx.x & 31;
  float vx = vec[e*3], vy = vec[e*3+1], vz = vec[e*3+2];
  float r  = sqrtf(vx*vx + vy*vy + vz*vz);
  float inv = __builtin_amdgcn_rcpf(r);
  float ux = vx*inv, uy = vy*inv, uz = vz*inv;
  if (lane == 0) {
    float d2 = r*r, d6 = d2*d2*d2, d7 = d6*r, d8 = d7*r;
    float ev = 1.0f - 28.0f*d6 + 48.0f*d7 - 21.0f*d8;
    env[e] = (r < 1.0f) ? ev : 0.0f;
  }
  const float s3 = 1.7320508075688772f, s5 = 2.23606797749979f, s15 = 3.872983346207417f;
  const float s7 = 2.6457513110645907f, s70 = 8.366600265340756f;
  const float s105 = 10.246950765959598f, s42 = 6.48074069840786f;
  float Y[16];
  Y[0]  = 1.0f;
  Y[1]  = s3*ux; Y[2] = s3*uy; Y[3] = s3*uz;
  Y[4]  = s15*ux*uy; Y[5] = s15*uy*uz; Y[6] = 0.5f*s5*(3.0f*uz*uz - 1.0f);
  Y[7]  = s15*ux*uz; Y[8] = 0.5f*s15*(ux*ux - uy*uy);
  Y[9]  = 0.25f*s70*uy*(3.0f*ux*ux - uy*uy);
  Y[10] = s105*ux*uy*uz;
  Y[11] = 0.25f*s42*uy*(5.0f*uz*uz - 1.0f);
  Y[12] = 0.5f*s7*uz*(5.0f*uz*uz - 3.0f);
  Y[13] = 0.25f*s42*ux*(5.0f*uz*uz - 1.0f);
  Y[14] = 0.5f*s105*uz*(ux*ux - uy*uy);
  Y[15] = 0.25f*s70*ux*(ux*ux - 3.0f*uy*uy);

  const float invs = 0.08838834764831845f; // 1/sqrt(128)
  float w0 = 0.0f, w1 = 0.0f;
  for (int i = 0; i < 128; ++i) {
    float xi = x[(size_t)e*128 + i];
    if ((i >> 2) == lane) xbf[(size_t)e*128 + i] = f2bfu(xi); // lane owns 4 elems
    w0 += xi * W_w[i*64 + lane];
    w1 += xi * W_w[i*64 + lane + 32];
  }
  w0 *= invs; w1 *= invs;
  int s = senders[e];
  float* b0 = nodeWY + (size_t)s*1024 + lane*16;
  float* b1 = b0 + 32*16;
#pragma unroll
  for (int k = 0; k < 16; ++k) {
    atomicAdd(b0 + k, w0 * Y[k]);
    atomicAdd(b1 + k, w1 * Y[k]);
  }
}

// ---------------- tensor product paths -------------------------------------
template <int L1, int L2, int L3>
__device__ inline void tp_path(const float* __restrict__ cgp, const float* __restrict__ wy,
                               const float* __restrict__ v, float* __restrict__ acc) {
  const int YO = (L1 == 0) ? 0 : (L1 == 1) ? 1 : (L1 == 2) ? 4 : 9;
  const int VO = (L2 == 0) ? 0 : (L2 == 1) ? 1 : 4;
#pragma unroll
  for (int i = 0; i < 2*L1 + 1; ++i)
#pragma unroll
    for (int j = 0; j < 2*L2 + 1; ++j) {
      float wv = wy[YO + i] * v[VO + j];
#pragma unroll
      for (int k = 0; k < 2*L3 + 1; ++k)
        acc[k] += cgp[(i*(2*L2+1) + j)*(2*L3+1) + k] * wv;
    }
}

__global__ void tp_kernel(const float* __restrict__ cg, const float* __restrict__ nodeWY,
                          const float* __restrict__ V, const int* __restrict__ senders,
                          unsigned short* __restrict__ hS, unsigned short* __restrict__ V1t,
                          unsigned short* __restrict__ V2t) {
  __shared__ float scg[CG_TOTAL];
  for (int t = threadIdx.x; t < CG_TOTAL; t += blockDim.x) scg[t] = cg[t];
  __syncthreads();
  int e    = blockIdx.x * 8 + (threadIdx.x >> 5);
  int lane = threadIdx.x & 31;
  int s    = senders[e];
  for (int rep = 0; rep < 2; ++rep) {
    int u = lane + rep * 32;
    float wy[16];
    const float4* pw = (const float4*)(nodeWY + (size_t)s*1024 + u*16);
#pragma unroll
    for (int q = 0; q < 4; ++q) {
      float4 w4 = pw[q];
      wy[4*q+0] = w4.x * 0.25f; wy[4*q+1] = w4.y * 0.25f;
      wy[4*q+2] = w4.z * 0.25f; wy[4*q+3] = w4.w * 0.25f;
    }
    float vv[9];
#pragma unroll
    for (int j = 0; j < 9; ++j) vv[j] = V[((size_t)e*64 + u)*9 + j];
    float outS[3] = {0.f, 0.f, 0.f};
    float o1[5][3] = {};
    float o2[5][5] = {};
    tp_path<0,0,0>(scg + 0,   wy, vv, &outS[0]);
    tp_path<0,1,1>(scg + 1,   wy, vv, o1[0]);
    tp_path<0,2,2>(scg + 10,  wy, vv, o2[0]);
    tp_path<1,0,1>(scg + 35,  wy, vv, o1[1]);
    tp_path<1,1,0>(scg + 44,  wy, vv, &outS[1]);
    tp_path<1,1,2>(scg + 53,  wy, vv, o2[1]);
    tp_path<1,2,1>(scg + 98,  wy, vv, o1[2]);
    tp_path<2,0,2>(scg + 143, wy, vv, o2[2]);
    tp_path<2,1,1>(scg + 168, wy, vv, o1[3]);
    tp_path<2,2,0>(scg + 213, wy, vv, &outS[2]);
    tp_path<2,2,2>(scg + 238, wy, vv, o2[3]);
    tp_path<3,1,2>(scg + 363, wy, vv, o2[4]);
    tp_path<3,2,1>(scg + 468, wy, vv, o1[4]);
#pragma unroll
    for (int p = 0; p < 3; ++p) hS[(size_t)e*192 + p*64 + u] = f2bfu(outS[p]);
#pragma unroll
    for (int p = 0; p < 5; ++p)
#pragma unroll
      for (int k = 0; k < 3; ++k)
        V1t[((size_t)e*3 + k)*320 + p*64 + u] = f2bfu(o1[p][k]);
#pragma unroll
    for (int p = 0; p < 5; ++p)
#pragma unroll
      for (int k = 0; k < 5; ++k)
        V2t[((size_t)e*5 + k)*320 + p*64 + u] = f2bfu(o2[p][k]);
  }
}

// ---------------- fused 3-layer MLP (bf16 WMMA, async A-tile staging) ------
__device__ inline float silu(float h) {
  return h * __builtin_amdgcn_rcpf(1.0f + __expf(-h));
}

__global__ void __launch_bounds__(128) mlp_kernel(const unsigned short* __restrict__ xbf,
    const unsigned short* __restrict__ hS, const unsigned short* __restrict__ W0t,
    const unsigned short* __restrict__ W1t, const unsigned short* __restrict__ W2t,
    const float* __restrict__ env, float* __restrict__ out) {
  __shared__ unsigned short Ab[16 * 320];
  __shared__ unsigned short Hb[16 * 64];
  int tile = blockIdx.x;
  int t    = threadIdx.x;

  // async-stage A tile: row m = [ xbf[e,0:128) | hS[e,0:192) ] as 40 x 16B chunks
  {
    unsigned ldsbase = lds_off_of(Ab);
    for (int idx = t; idx < 640; idx += 128) {
      int m = idx / 40, c = idx % 40;
      int e = tile * 16 + m;
      unsigned long long g = (c < 16)
          ? (unsigned long long)(xbf + (size_t)e * 128 + c * 8)
          : (unsigned long long)(hS  + (size_t)e * 192 + (c - 16) * 8);
      async_cp16(ldsbase + (unsigned)(m * 640 + c * 16), g);
    }
    async_wait0();
  }
  __syncthreads();

  int wv = t >> 5, lane = t & 31;
  int half = lane >> 4, nn = lane & 15;
  int n0 = wv * 16;

  // GEMM1: [x|scal](16x320) x W0(320x64)   (scales pre-folded into W*t)
  v8f c = {0.f,0.f,0.f,0.f,0.f,0.f,0.f,0.f};
  const unsigned short* arow = Ab + nn * 320;
  const unsigned short* bcol = W0t + (size_t)(n0 + nn) * 320;
#pragma unroll
  for (int kc = 0; kc < 10; ++kc) {
    v16bf a = loadA_row(arow, kc * 32, half);
    v16bf b = loadB_col(bcol, kc * 32, half);
    c = __builtin_amdgcn_wmma_f32_16x16x32_bf16(false, a, false, b, (short)0, c, false, false);
  }
#pragma unroll
  for (int r = 0; r < 8; ++r) {
    int m = r + 8 * half;
    Hb[m * 64 + n0 + nn] = f2bfu(silu(c[r]));
  }
  __syncthreads();

  // GEMM2: h1(16x64) x W1(64x64)
  v8f c2 = {0.f,0.f,0.f,0.f,0.f,0.f,0.f,0.f};
  const unsigned short* hrow = Hb + nn * 64;
  const unsigned short* b1col = W1t + (size_t)(n0 + nn) * 64;
#pragma unroll
  for (int kc = 0; kc < 2; ++kc) {
    v16bf a = loadA_row(hrow, kc * 32, half);
    v16bf b = loadB_col(b1col, kc * 32, half);
    c2 = __builtin_amdgcn_wmma_f32_16x16x32_bf16(false, a, false, b, (short)0, c2, false, false);
  }
  __syncthreads();
#pragma unroll
  for (int r = 0; r < 8; ++r) {
    int m = r + 8 * half;
    Hb[m * 64 + n0 + nn] = f2bfu(silu(c2[r]));
  }
  __syncthreads();

  // GEMM3: h2(16x64) x W2(64x64), epilogue * envelope(r)
  v8f c3 = {0.f,0.f,0.f,0.f,0.f,0.f,0.f,0.f};
  const unsigned short* b2col = W2t + (size_t)(n0 + nn) * 64;
#pragma unroll
  for (int kc = 0; kc < 2; ++kc) {
    v16bf a = loadA_row(hrow, kc * 32, half);
    v16bf b = loadB_col(b2col, kc * 32, half);
    c3 = __builtin_amdgcn_wmma_f32_16x16x32_bf16(false, a, false, b, (short)0, c3, false, false);
  }
#pragma unroll
  for (int r = 0; r < 8; ++r) {
    int m = r + 8 * half;
    int e = tile * 16 + m;
    out[(size_t)e*64 + n0 + nn] = env[e] * c3[r];
  }
}

// ---------------- V-channel output GEMM (bf16 WMMA, async staging) ---------
__global__ void __launch_bounds__(128) vout_kernel(const unsigned short* __restrict__ At,
    const unsigned short* __restrict__ Wt, float* __restrict__ outB, int Kdiv, int koff) {
  __shared__ unsigned short Ab[16 * 320];
  int tile = blockIdx.x;
  int t    = threadIdx.x;
  {
    unsigned long long gbase = (unsigned long long)(At + (size_t)tile * 16 * 320);
    unsigned ldsbase = lds_off_of(Ab);
    for (int idx = t; idx < 640; idx += 128)
      async_cp16(ldsbase + (unsigned)(idx * 16), gbase + (unsigned long long)idx * 16ull);
    async_wait0();
  }
  __syncthreads();

  int wv = t >> 5, lane = t & 31;
  int half = lane >> 4, nn = lane & 15;
  int n0 = wv * 16;
  v8f c = {0.f,0.f,0.f,0.f,0.f,0.f,0.f,0.f};
  const unsigned short* arow = Ab + nn * 320;
  const unsigned short* bcol = Wt + (size_t)(n0 + nn) * 320;
#pragma unroll
  for (int kc = 0; kc < 10; ++kc) {
    v16bf a = loadA_row(arow, kc * 32, half);
    v16bf b = loadB_col(bcol, kc * 32, half);
    c = __builtin_amdgcn_wmma_f32_16x16x32_bf16(false, a, false, b, (short)0, c, false, false);
  }
#pragma unroll
  for (int r = 0; r < 8; ++r) {
    int m  = r + 8 * half;
    int rr = tile * 16 + m;
    int e  = rr / Kdiv;
    int k  = rr - e * Kdiv;
    outB[(size_t)e*512 + koff + (n0 + nn)*Kdiv + k] = c[r];
  }
}

// ---------------- launcher --------------------------------------------------
extern "C" void kernel_launch(void* const* d_in, const int* in_sizes, int n_in,
                              void* d_out, int out_size, void* d_ws, size_t ws_size,
                              hipStream_t stream) {
  const float* vectors = (const float*)d_in[0];
  const float* x       = (const float*)d_in[1];
  const float* V       = (const float*)d_in[2];
  const int*   senders = (const int*)d_in[3];
  const float* W_w     = (const float*)d_in[4];
  const float* W0      = (const float*)d_in[5];
  const float* W1      = (const float*)d_in[6];
  const float* W2      = (const float*)d_in[7];
  const float* Wo1     = (const float*)d_in[8];
  const float* Wo2     = (const float*)d_in[9];
  float* out = (float*)d_out;
  const int E = N_EDGES, NN = N_NODES;

  char* ws = (char*)d_ws;
  float* cg           = (float*)ws;          ws += 4096;
  float* env          = (float*)ws;          ws += (size_t)E * 4;
  float* nodeWY       = (float*)ws;          ws += (size_t)NN * 1024 * 4;
  unsigned short* xbf  = (unsigned short*)ws; ws += (size_t)E * 128 * 2;
  unsigned short* hS   = (unsigned short*)ws; ws += (size_t)E * 192 * 2;
  unsigned short* V1t  = (unsigned short*)ws; ws += (size_t)E * 3 * 320 * 2;
  unsigned short* V2t  = (unsigned short*)ws; ws += (size_t)E * 5 * 320 * 2;
  unsigned short* W0t  = (unsigned short*)ws; ws += (size_t)64 * 320 * 2;
  unsigned short* W1t  = (unsigned short*)ws; ws += (size_t)64 * 64 * 2;
  unsigned short* W2t  = (unsigned short*)ws; ws += (size_t)64 * 64 * 2;
  unsigned short* Wo1t = (unsigned short*)ws; ws += (size_t)64 * 320 * 2;
  unsigned short* Wo2t = (unsigned short*)ws; ws += (size_t)64 * 320 * 2;

  const float s320 = 0.05590169943749474f; // 1/sqrt(320)
  const float s64  = 0.125f;               // 1/sqrt(64)

  cg_init_kernel<<<1, 576, 0, stream>>>(cg);
  zero_kernel<<<(NN * 1024 + 255) / 256, 256, 0, stream>>>(nodeWY, NN * 1024);
  prep_w_kernel<<<80, 256, 0, stream>>>(W0,  W0t,  320, s320);
  prep_w_kernel<<<16, 256, 0, stream>>>(W1,  W1t,  64,  s64);
  prep_w_kernel<<<16, 256, 0, stream>>>(W2,  W2t,  64,  s64);
  prep_w_kernel<<<80, 256, 0, stream>>>(Wo1, Wo1t, 320, s320);
  prep_w_kernel<<<80, 256, 0, stream>>>(Wo2, Wo2t, 320, s320);
  edge_pre_kernel<<<E / 8, 256, 0, stream>>>(vectors, x, senders, W_w, nodeWY, env, xbf);
  tp_kernel<<<E / 8, 256, 0, stream>>>(cg, nodeWY, V, senders, hS, V1t, V2t);
  mlp_kernel<<<E / 16, 128, 0, stream>>>(xbf, hS, W0t, W1t, W2t, env, out);
  vout_kernel<<<(E * 3) / 16, 128, 0, stream>>>(V1t, Wo1t, out + (size_t)E * 64, 3, 0);
  vout_kernel<<<(E * 5) / 16, 128, 0, stream>>>(V2t, Wo2t, out + (size_t)E * 64, 5, 192);
}
